// CategoryBranch_68212670595109
// MI455X (gfx1250) — matrix operand order repaired and checked by reading, not compiled, not run on hardware
//
#include <hip/hip_runtime.h>
#include <hip/hip_bf16.h>

// Problem constants (match reference)
#define B_    64
#define TCTX  512
#define D_    1024
#define K_    10
#define TOUT  128

typedef __attribute__((ext_vector_type(8)))  float   v8f;
typedef __attribute__((ext_vector_type(8)))  __bf16  v8bf;
typedef __attribute__((ext_vector_type(16))) __bf16  v16bf;

// ---- helpers ----------------------------------------------------------------

__device__ __forceinline__ __bf16 f2bf(float f) {
  // round-to-nearest-even truncation to bf16 (storage-only, no bf16 arithmetic)
  unsigned u = __builtin_bit_cast(unsigned, f);
  unsigned r = u + 0x7FFFu + ((u >> 16) & 1u);
  unsigned short h = (unsigned short)(r >> 16);
  return __builtin_bit_cast(__bf16, h);
}

// A-matrix fragment (16-bit, 16x32, row-major source, row pointer given):
//   lanes 0-15 : K = k0+0..7   (v0..3) and k0+16..23 (v4..7)
//   lanes 16-31: K = k0+8..15  (v0..3) and k0+24..31 (v4..7)
__device__ __forceinline__ v16bf load_a_frag(const __bf16* __restrict__ row,
                                             int k0, int half) {
  int kb = k0 + half * 8;
  v8bf lo = *(const v8bf*)(row + kb);        // 16B load
  v8bf hi = *(const v8bf*)(row + kb + 16);   // 16B load
  return __builtin_shufflevector(lo, hi, 0,1,2,3,4,5,6,7,8,9,10,11,12,13,14,15);
}

// B-matrix fragment (16-bit, 32x16, column n = lane&15, source row = weight row
// for column n since B = W^T):
//   lanes 0-15 : K = k0+0..7 and k0+16..23 ; lanes 16-31: K = k0+8..15 and k0+24..31
__device__ __forceinline__ v16bf load_b_frag(const __bf16* __restrict__ wrow,
                                             int k0, int half) {
  int kb = k0 + half * 8;
  v8bf lo = *(const v8bf*)(wrow + kb);
  v8bf hi = *(const v8bf*)(wrow + kb + 16);
  return __builtin_shufflevector(lo, hi, 0,1,2,3,4,5,6,7,8,9,10,11,12,13,14,15);
}

#define WMMA_BF16(A, Bf, C) \
  __builtin_amdgcn_wmma_f32_16x16x32_bf16(false, (A), false, (Bf), (short)0, (C), false, false)

// ---- prologue kernels -------------------------------------------------------

__global__ void cvt_f32_bf16_kernel(const float* __restrict__ src,
                                    __bf16* __restrict__ dst, int n) {
  int i = blockIdx.x * blockDim.x + threadIdx.x;
  if (i < n) dst[i] = f2bf(src[i]);
}

__global__ void init_h_kernel(const float* __restrict__ h0,
                              float* __restrict__ h,
                              __bf16* __restrict__ hb, int n) {
  int i = blockIdx.x * blockDim.x + threadIdx.x;
  if (i < n) { float v = h0[i]; h[i] = v; hb[i] = f2bf(v); }
}

// ---- step kernel 1: p = h @ W_att, then Graves phi --------------------------
// one block per batch row; 256 threads
__global__ __launch_bounds__(256) void attn_phi_kernel(
    const float* __restrict__ h,      // [B, D]
    const float* __restrict__ W_att,  // [D, 3K] row-major
    float* __restrict__ phi)          // [B, TCTX]
{
  __shared__ float red[30 * 256];
  __shared__ float prm[30];           // exp(a) | exp(b) | exp(k)  (alpha,beta,kappa)
  const int b = blockIdx.x, tid = threadIdx.x;

  float pc[30];
#pragma unroll
  for (int c = 0; c < 30; ++c) pc[c] = 0.f;
  for (int d = tid; d < D_; d += 256) {
    float hv = h[b * D_ + d];
    const float* w = W_att + d * 30;
#pragma unroll
    for (int c = 0; c < 30; ++c) pc[c] += hv * w[c];
  }
#pragma unroll
  for (int c = 0; c < 30; ++c) red[c * 256 + tid] = pc[c];
  __syncthreads();
  if (tid < 30) {
    float s = 0.f;
    for (int i = 0; i < 256; ++i) s += red[tid * 256 + i];
    prm[tid] = __expf(s);             // alpha = e^a, beta = e^b, kappa = 0 + e^k
  }
  __syncthreads();
  for (int t = tid; t < TCTX; t += 256) {
    float s = 0.f;
#pragma unroll
    for (int k = 0; k < K_; ++k) {
      float dist = prm[20 + k] - (float)t;
      s += prm[k] * __expf(-prm[10 + k] * dist * dist);
    }
    phi[b * TCTX + t] = s;
  }
}

// ---- step kernel 2: attended[b,d] = sum_t phi[b,t]*context[b,t,d] -----------
// grid (D/128, B), 128 threads; context streams from L2 (whole tensor fits)
__global__ __launch_bounds__(128) void attend_kernel(
    const float* __restrict__ phi,     // [B, TCTX]
    const float* __restrict__ ctx,     // [B, TCTX, D]
    float* __restrict__ att,           // [B, D] f32
    __bf16* __restrict__ att_bf)       // [B, D] bf16 (GEMM operand)
{
  __shared__ float ph[TCTX];
  const int b = blockIdx.y;
  const int d = blockIdx.x * 128 + threadIdx.x;
  for (int t = threadIdx.x; t < TCTX; t += 128) ph[t] = phi[b * TCTX + t];
  __syncthreads();
  const float* cb = ctx + (size_t)b * TCTX * D_ + d;
  float acc = 0.f;
#pragma unroll 4
  for (int t = 0; t < TCTX; ++t) acc += ph[t] * cb[(size_t)t * D_];
  att[b * D_ + d] = acc;
  att_bf[b * D_ + d] = f2bf(acc);
}

// ---- step kernel 3: fused GRU cell ------------------------------------------
// gi = x @ W_ih^T, gh = h @ W_hh^T, gates, h_new.  64 blocks (16 cols each),
// 4 waves per block (16 batch rows each).  6 WMMA accumulators per wave,
// elementwise-aligned for the gate epilogue.
__global__ __launch_bounds__(128) void gru_step_kernel(
    const __bf16* __restrict__ x_bf,    // attended bf16 [B, D]
    const __bf16* __restrict__ h_bf,    // hidden bf16   [B, D]
    const __bf16* __restrict__ Wih,     // [3D, D] bf16
    const __bf16* __restrict__ Whh,     // [3D, D] bf16
    const float*  __restrict__ h_cur,   // [B, D] f32
    float* __restrict__ h_next,         // [B, D] f32
    __bf16* __restrict__ h_next_bf,     // [B, D] bf16
    float* __restrict__ out,            // [B, TOUT, D]
    int t)
{
  const int lane = threadIdx.x & 31;
  const int wave = threadIdx.x >> 5;
  const int l16  = lane & 15;
  const int half = lane >> 4;
  const int j0   = blockIdx.x * 16;   // output-column tile
  const int m0   = wave * 16;         // batch-row tile

  v8f acc_ir = {}, acc_iz = {}, acc_in = {};
  v8f acc_hr = {}, acc_hz = {}, acc_hn = {};

  const __bf16* xrow = x_bf + (size_t)(m0 + l16) * D_;
  const __bf16* hrow = h_bf + (size_t)(m0 + l16) * D_;
  const __bf16* wir  = Wih + (size_t)(0 * D_ + j0 + l16) * D_;
  const __bf16* wiz  = Wih + (size_t)(1 * D_ + j0 + l16) * D_;
  const __bf16* win  = Wih + (size_t)(2 * D_ + j0 + l16) * D_;
  const __bf16* whr  = Whh + (size_t)(0 * D_ + j0 + l16) * D_;
  const __bf16* whz  = Whh + (size_t)(1 * D_ + j0 + l16) * D_;
  const __bf16* whn  = Whh + (size_t)(2 * D_ + j0 + l16) * D_;

  for (int k0 = 0; k0 < D_; k0 += 32) {
    v16bf ax = load_a_frag(xrow, k0, half);
    v16bf ah = load_a_frag(hrow, k0, half);
    v16bf br = load_b_frag(wir, k0, half);
    v16bf bz = load_b_frag(wiz, k0, half);
    v16bf bn = load_b_frag(win, k0, half);
    v16bf cr = load_b_frag(whr, k0, half);
    v16bf cz = load_b_frag(whz, k0, half);
    v16bf cn = load_b_frag(whn, k0, half);
    acc_ir = WMMA_BF16(ax, br, acc_ir);
    acc_iz = WMMA_BF16(ax, bz, acc_iz);
    acc_in = WMMA_BF16(ax, bn, acc_in);
    acc_hr = WMMA_BF16(ah, cr, acc_hr);
    acc_hz = WMMA_BF16(ah, cz, acc_hz);
    acc_hn = WMMA_BF16(ah, cn, acc_hn);
  }

  // epilogue: C/D layout -> element (m = r + 8*half, n = l16)
  const int j = j0 + l16;
#pragma unroll
  for (int r = 0; r < 8; ++r) {
    const int b = m0 + 8 * half + r;
    float rr = acc_ir[r] + acc_hr[r];
    float zz = acc_iz[r] + acc_hz[r];
    rr = 1.f / (1.f + __expf(-rr));
    zz = 1.f / (1.f + __expf(-zz));
    float nn = tanhf(acc_in[r] + rr * acc_hn[r]);
    float hv = h_cur[b * D_ + j];
    float hnew = (1.f - zz) * nn + zz * hv;
    h_next[b * D_ + j] = hnew;
    h_next_bf[b * D_ + j] = f2bf(hnew);
    out[((size_t)b * TOUT + t) * D_ + j] = hnew;
  }
}

// ---- host driver ------------------------------------------------------------

extern "C" void kernel_launch(void* const* d_in, const int* in_sizes, int n_in,
                              void* d_out, int out_size, void* d_ws, size_t ws_size,
                              hipStream_t stream) {
  const float* context = (const float*)d_in[0];  // [B, TCTX, D]
  const float* hidden0 = (const float*)d_in[1];  // [B, D]
  const float* W_att   = (const float*)d_in[2];  // [D, 3K]
  const float* W_ih    = (const float*)d_in[3];  // [3D, D]
  const float* W_hh    = (const float*)d_in[4];  // [3D, D]
  float* out = (float*)d_out;                    // [B, TOUT, D]

  // workspace carve-up (256B aligned)
  char* ws = (char*)d_ws;
  size_t off = 0;
  auto carve = [&](size_t bytes) -> void* {
    void* p = ws + off;
    off = (off + bytes + 255) & ~(size_t)255;
    return p;
  };
  const int NW   = 3 * D_ * D_;          // 3,145,728 weight elements
  __bf16* Wih_bf = (__bf16*)carve((size_t)NW * 2);
  __bf16* Whh_bf = (__bf16*)carve((size_t)NW * 2);
  float*  h_a    = (float*) carve((size_t)B_ * D_ * 4);
  float*  h_b    = (float*) carve((size_t)B_ * D_ * 4);
  __bf16* hb_a   = (__bf16*)carve((size_t)B_ * D_ * 2);
  __bf16* hb_b   = (__bf16*)carve((size_t)B_ * D_ * 2);
  float*  att    = (float*) carve((size_t)B_ * D_ * 4);
  __bf16* att_bf = (__bf16*)carve((size_t)B_ * D_ * 2);
  float*  phi    = (float*) carve((size_t)B_ * TCTX * 4);

  // one-time bf16 weight conversion (deterministic, re-done every call)
  cvt_f32_bf16_kernel<<<(NW + 255) / 256, 256, 0, stream>>>(W_ih, Wih_bf, NW);
  cvt_f32_bf16_kernel<<<(NW + 255) / 256, 256, 0, stream>>>(W_hh, Whh_bf, NW);
  init_h_kernel<<<(B_ * D_ + 255) / 256, 256, 0, stream>>>(hidden0, h_a, hb_a, B_ * D_);

  float*  h_cur = h_a,  *h_nxt = h_b;
  __bf16* hb_cur = hb_a, *hb_nxt = hb_b;

  for (int t = 0; t < TOUT; ++t) {
    attn_phi_kernel<<<B_, 256, 0, stream>>>(h_cur, W_att, phi);
    attend_kernel<<<dim3(D_ / 128, B_), 128, 0, stream>>>(phi, context, att, att_bf);
    gru_step_kernel<<<D_ / 16, 128, 0, stream>>>(att_bf, hb_cur, Wih_bf, Whh_bf,
                                                 h_cur, h_nxt, hb_nxt, out, t);
    // ping-pong hidden state
    float* tf = h_cur;  h_cur = h_nxt;  h_nxt = tf;
    __bf16* tb = hb_cur; hb_cur = hb_nxt; hb_nxt = tb;
  }
  (void)in_sizes; (void)n_in; (void)out_size; (void)ws_size;
}
